// PackedExperts_73315091743524
// MI455X (gfx1250) — matrix operand order repaired and checked by reading, not compiled, not run on hardware
//
#include <hip/hip_runtime.h>
#include <hip/hip_bf16.h>

#define NUM_EXPERTS 16
#define HIDDEN 2048
#define INTER  4096
#define TOP_K  2
#define BATCH  1024
#define NASSIGN (BATCH * TOP_K)

#define BM 32
#define BN 128
#define BK 64            // real K per staging round (2 WMMA sub-steps of 32)
#define KPAD 8           // pad to 72 halves = 144B pitch (multiple of 16B)
#define NTHREADS 256

typedef __attribute__((ext_vector_type(16))) __bf16 v16bf;
typedef __attribute__((ext_vector_type(8)))  __bf16 v8bf;
typedef __attribute__((ext_vector_type(4)))  __bf16 v4bf;
typedef __attribute__((ext_vector_type(8)))  float  v8f;
typedef __attribute__((ext_vector_type(4)))  float  v4f;

__device__ __forceinline__ void split_bf16(float x, __bf16& h, __bf16& l) {
    h = (__bf16)x;
    l = (__bf16)(x - (float)h);
}

__device__ __forceinline__ v16bf join16(v8bf lo, v8bf hi) {
    return __builtin_shufflevector(lo, hi, 0,1,2,3,4,5,6,7,8,9,10,11,12,13,14,15);
}

#define WMMA_BF16(A, B, C) \
    __builtin_amdgcn_wmma_f32_16x16x32_bf16(false, (A), false, (B), (short)0, (C), false, false)

// CDNA5 async global->LDS copy (ASYNCcnt-tracked DMA, no VGPR data path)
__device__ __forceinline__ unsigned lds_off(const void* p) {
    return (unsigned)(unsigned long long)p;   // low 32 bits of generic addr = LDS offset
}
__device__ __forceinline__ void async_copy_b128(unsigned dst_lds, const void* src_global) {
    asm volatile("global_load_async_to_lds_b128 %0, %1, off"
                 :: "v"(dst_lds), "v"(src_global) : "memory");
}
__device__ __forceinline__ void wait_async0() {
    asm volatile("s_wait_asynccnt 0" ::: "memory");
}

__device__ __forceinline__ int imin(int a, int b) { return a < b ? a : b; }

// ---------------- routing ----------------

__global__ void k_route_init(int* counts, int* cursor) {
    int i = threadIdx.x;
    if (i < NUM_EXPERTS) { counts[i] = 0; cursor[i] = 0; }
}

__global__ void k_route_count(const int* __restrict__ topk_idx, int* counts) {
    int i = blockIdx.x * blockDim.x + threadIdx.x;
    if (i < NASSIGN) atomicAdd(&counts[topk_idx[i] & (NUM_EXPERTS - 1)], 1);
}

__global__ void k_route_scan(const int* __restrict__ counts, int* offsets, int* cursor) {
    if (threadIdx.x == 0) {
        int acc = 0;
        for (int e = 0; e < NUM_EXPERTS; ++e) {
            offsets[e] = acc; cursor[e] = acc; acc += counts[e];
        }
        offsets[NUM_EXPERTS] = acc;
    }
}

__global__ void k_route_scatter(const int* __restrict__ topk_idx, int* cursor,
                                int* tok_idx, int* pos_of) {
    int i = blockIdx.x * blockDim.x + threadIdx.x;
    if (i < NASSIGN) {
        int e = topk_idx[i] & (NUM_EXPERTS - 1);
        int pos = atomicAdd(&cursor[e], 1);
        tok_idx[pos] = i >> 1;
        pos_of[i] = pos;
    }
}

// stage one BK x BN fp32 weight tile as transposed bf16 hi/lo planes.
// thread layout: cols cb=(t&31)*4 (512B-coalesced row reads), k-groups kb=4*(wave+8g)
__device__ __forceinline__ void stage_B(const float* __restrict__ wrow_base, int ld,
                                        int k0, int n0, int wave, int laneq,
                                        __bf16 (*BThi)[BK + KPAD], __bf16 (*BTlo)[BK + KPAD])
{
    int cb = laneq * 4;
    #pragma unroll
    for (int g = 0; g < 2; ++g) {
        int kb = (wave + 8 * g) * 4;
        v4f r[4];
        #pragma unroll
        for (int i = 0; i < 4; ++i)
            r[i] = *(const v4f*)(wrow_base + (size_t)(k0 + kb + i) * ld + n0 + cb);
        #pragma unroll
        for (int j = 0; j < 4; ++j) {           // transpose 4x4 in registers
            v4bf th, tl;
            #pragma unroll
            for (int i = 0; i < 4; ++i) {
                __bf16 h, l; split_bf16(r[i][j], h, l);
                th[i] = h; tl[i] = l;
            }
            *(v4bf*)&BThi[cb + j][kb] = th;
            *(v4bf*)&BTlo[cb + j][kb] = tl;
        }
    }
}

// ---------------- GEMM1: h = relu(X_e @ W1[e] + b1[e]) ----------------

__global__ __launch_bounds__(NTHREADS) void k_gemm1(
    const float* __restrict__ tokens, const float* __restrict__ w1,
    const float* __restrict__ b1, const int* __restrict__ offsets,
    const int* __restrict__ tok_idx,
    __bf16* __restrict__ h_hi, __bf16* __restrict__ h_lo)
{
    int e    = blockIdx.z;
    int base = offsets[e];
    int cnt  = offsets[e + 1] - base;
    int m0   = blockIdx.y * BM;
    if (m0 >= cnt) return;
    int n0   = blockIdx.x * BN;

    __shared__ __align__(16) __bf16 Ahi[BM][BK + KPAD];
    __shared__ __align__(16) __bf16 Alo[BM][BK + KPAD];
    __shared__ __align__(16) __bf16 BThi[BN][BK + KPAD];
    __shared__ __align__(16) __bf16 BTlo[BN][BK + KPAD];

    int tid  = threadIdx.x;
    int lane = tid & 31;
    int wave = tid >> 5;
    int laneq = tid & 31;
    int lrow = lane & 15;
    int kA   = (lane < 16) ? 0 : 8;
    int kB   = (lane < 16) ? 0 : 8;

    // A staging: thread t -> row t/8, 8 floats at col 8*(t%8)
    int a_r = tid >> 3;
    int a_c = (tid & 7) * 8;
    int a_rl = imin(a_r, cnt - 1 - m0);
    const float* a_src = tokens + (size_t)tok_idx[base + m0 + a_rl] * HIDDEN + a_c;

    const float* w1e = w1 + (size_t)e * HIDDEN * INTER;

    v8f acc0 = {}; v8f acc1 = {};

    for (int k0 = 0; k0 < HIDDEN; k0 += BK) {
        // stage A (fp32 -> bf16 hi/lo, packed b64 stores)
        #pragma unroll
        for (int q = 0; q < 2; ++q) {
            v4f av = *(const v4f*)(a_src + k0 + q * 4);
            v4bf th, tl;
            #pragma unroll
            for (int j = 0; j < 4; ++j) {
                __bf16 h, l; split_bf16(av[j], h, l);
                th[j] = h; tl[j] = l;
            }
            *(v4bf*)&Ahi[a_r][a_c + q * 4] = th;
            *(v4bf*)&Alo[a_r][a_c + q * 4] = tl;
        }
        stage_B(w1e, INTER, k0, n0, wave, laneq, BThi, BTlo);
        __syncthreads();

        #pragma unroll
        for (int s = 0; s < 2; ++s) {
            int ko = s * 32;
            const __bf16* a0h = &Ahi[lrow][ko + kA];
            const __bf16* a0l = &Alo[lrow][ko + kA];
            const __bf16* a1h = &Ahi[16 + lrow][ko + kA];
            const __bf16* a1l = &Alo[16 + lrow][ko + kA];
            v16bf fA0h = join16(*(const v8bf*)a0h, *(const v8bf*)(a0h + 16));
            v16bf fA0l = join16(*(const v8bf*)a0l, *(const v8bf*)(a0l + 16));
            v16bf fA1h = join16(*(const v8bf*)a1h, *(const v8bf*)(a1h + 16));
            v16bf fA1l = join16(*(const v8bf*)a1l, *(const v8bf*)(a1l + 16));
            const __bf16* bh = &BThi[wave * 16 + lrow][ko + kB];
            const __bf16* bl = &BTlo[wave * 16 + lrow][ko + kB];
            v16bf fBh = join16(*(const v8bf*)bh, *(const v8bf*)(bh + 16));
            v16bf fBl = join16(*(const v8bf*)bl, *(const v8bf*)(bl + 16));

            acc0 = WMMA_BF16(fA0h, fBh, acc0);
            acc0 = WMMA_BF16(fA0h, fBl, acc0);
            acc0 = WMMA_BF16(fA0l, fBh, acc0);
            acc1 = WMMA_BF16(fA1h, fBh, acc1);
            acc1 = WMMA_BF16(fA1h, fBl, acc1);
            acc1 = WMMA_BF16(fA1l, fBh, acc1);
        }
        __syncthreads();
    }

    int n = n0 + wave * 16 + lrow;
    float bias = b1[e * INTER + n];
    int mbase = (lane < 16) ? 0 : 8;
    #pragma unroll
    for (int r = 0; r < 8; ++r) {
        int m = r + mbase;
        if (m0 + m < cnt) {
            float v = acc0[r] + bias; v = v > 0.f ? v : 0.f;
            __bf16 h, l; split_bf16(v, h, l);
            size_t idx = (size_t)(base + m0 + m) * INTER + n;
            h_hi[idx] = h; h_lo[idx] = l;
        }
        int m2 = 16 + r + mbase;
        if (m0 + m2 < cnt) {
            float v = acc1[r] + bias; v = v > 0.f ? v : 0.f;
            __bf16 h, l; split_bf16(v, h, l);
            size_t idx = (size_t)(base + m0 + m2) * INTER + n;
            h_hi[idx] = h; h_lo[idx] = l;
        }
    }
}

// ---------------- GEMM2: pairs = h @ W2[e] + b2[e] ----------------
// A tile (already bf16, fragment-ready layout) staged via async global->LDS DMA.

__global__ __launch_bounds__(NTHREADS) void k_gemm2(
    const __bf16* __restrict__ h_hi, const __bf16* __restrict__ h_lo,
    const float* __restrict__ w2, const float* __restrict__ b2,
    const int* __restrict__ offsets, float* __restrict__ out_pairs)
{
    int e    = blockIdx.z;
    int base = offsets[e];
    int cnt  = offsets[e + 1] - base;
    int m0   = blockIdx.y * BM;
    if (m0 >= cnt) return;
    int n0   = blockIdx.x * BN;

    __shared__ __align__(16) __bf16 A2hi[BM][BK + KPAD];
    __shared__ __align__(16) __bf16 A2lo[BM][BK + KPAD];
    __shared__ __align__(16) __bf16 BThi[BN][BK + KPAD];
    __shared__ __align__(16) __bf16 BTlo[BN][BK + KPAD];

    int tid  = threadIdx.x;
    int lane = tid & 31;
    int wave = tid >> 5;
    int laneq = tid & 31;
    int lrow = lane & 15;
    int kA   = (lane < 16) ? 0 : 8;
    int kB   = (lane < 16) ? 0 : 8;

    const float* w2e = w2 + (size_t)e * INTER * HIDDEN;

    // async A staging: thread t copies 16B chunk (row rr, seg) of both planes
    int rr  = tid >> 3;
    int seg = tid & 7;
    int rloc = imin(rr, cnt - 1 - m0);
    const __bf16* srcH = h_hi + (size_t)(base + m0 + rloc) * INTER + seg * 8;
    const __bf16* srcL = h_lo + (size_t)(base + m0 + rloc) * INTER + seg * 8;
    unsigned dstH = lds_off(&A2hi[rr][seg * 8]);
    unsigned dstL = lds_off(&A2lo[rr][seg * 8]);

    v8f acc0 = {}; v8f acc1 = {};

    for (int k0 = 0; k0 < INTER; k0 += BK) {
        async_copy_b128(dstH, srcH + k0);     // DMA, overlaps B conversion below
        async_copy_b128(dstL, srcL + k0);
        stage_B(w2e, HIDDEN, k0, n0, wave, laneq, BThi, BTlo);
        wait_async0();
        __syncthreads();

        #pragma unroll
        for (int s = 0; s < 2; ++s) {
            int ko = s * 32;
            const __bf16* a0h = &A2hi[lrow][ko + kA];
            const __bf16* a0l = &A2lo[lrow][ko + kA];
            const __bf16* a1h = &A2hi[16 + lrow][ko + kA];
            const __bf16* a1l = &A2lo[16 + lrow][ko + kA];
            v16bf fA0h = join16(*(const v8bf*)a0h, *(const v8bf*)(a0h + 16));
            v16bf fA0l = join16(*(const v8bf*)a0l, *(const v8bf*)(a0l + 16));
            v16bf fA1h = join16(*(const v8bf*)a1h, *(const v8bf*)(a1h + 16));
            v16bf fA1l = join16(*(const v8bf*)a1l, *(const v8bf*)(a1l + 16));
            const __bf16* bh = &BThi[wave * 16 + lrow][ko + kB];
            const __bf16* bl = &BTlo[wave * 16 + lrow][ko + kB];
            v16bf fBh = join16(*(const v8bf*)bh, *(const v8bf*)(bh + 16));
            v16bf fBl = join16(*(const v8bf*)bl, *(const v8bf*)(bl + 16));

            acc0 = WMMA_BF16(fA0h, fBh, acc0);
            acc0 = WMMA_BF16(fA0h, fBl, acc0);
            acc0 = WMMA_BF16(fA0l, fBh, acc0);
            acc1 = WMMA_BF16(fA1h, fBh, acc1);
            acc1 = WMMA_BF16(fA1h, fBl, acc1);
            acc1 = WMMA_BF16(fA1l, fBh, acc1);
        }
        __syncthreads();
    }

    int n = n0 + wave * 16 + lrow;
    float bias = b2[e * HIDDEN + n];
    int mbase = (lane < 16) ? 0 : 8;
    #pragma unroll
    for (int r = 0; r < 8; ++r) {
        int m = r + mbase;
        if (m0 + m < cnt)
            out_pairs[(size_t)(base + m0 + m) * HIDDEN + n] = acc0[r] + bias;
        int m2 = 16 + r + mbase;
        if (m0 + m2 < cnt)
            out_pairs[(size_t)(base + m0 + m2) * HIDDEN + n] = acc1[r] + bias;
    }
}

// ---------------- combine (deterministic, no output atomics) ----------------

__global__ void k_combine(const float* __restrict__ out_pairs,
                          const int* __restrict__ pos_of,
                          const float* __restrict__ topk_w,
                          float* __restrict__ out)
{
    int i = blockIdx.x * blockDim.x + threadIdx.x;
    if (i >= BATCH * HIDDEN) return;
    int t = i / HIDDEN;
    int j = i - t * HIDDEN;
    float r = topk_w[t * TOP_K + 0] * out_pairs[(size_t)pos_of[t * TOP_K + 0] * HIDDEN + j]
            + topk_w[t * TOP_K + 1] * out_pairs[(size_t)pos_of[t * TOP_K + 1] * HIDDEN + j];
    out[i] = r;
}

// ---------------- launch ----------------

extern "C" void kernel_launch(void* const* d_in, const int* in_sizes, int n_in,
                              void* d_out, int out_size, void* d_ws, size_t ws_size,
                              hipStream_t stream) {
    const float* tokens   = (const float*)d_in[0];
    const int*   topk_idx = (const int*)  d_in[1];
    const float* topk_w   = (const float*)d_in[2];
    const float* w1       = (const float*)d_in[3];
    const float* b1       = (const float*)d_in[4];
    const float* w2       = (const float*)d_in[5];
    const float* b2       = (const float*)d_in[6];
    float* out = (float*)d_out;

    char* ws = (char*)d_ws;
    int* counts  = (int*)ws;
    int* offsets = counts + 16;
    int* cursor  = offsets + 17;
    int* tok_idx = cursor + 16;
    int* pos_of  = tok_idx + NASSIGN;
    __bf16* h_hi = (__bf16*)(ws + 32768);
    __bf16* h_lo = h_hi + (size_t)NASSIGN * INTER;
    float* out_pairs = (float*)(h_lo + (size_t)NASSIGN * INTER);

    k_route_init   <<<1, 64, 0, stream>>>(counts, cursor);
    k_route_count  <<<NASSIGN / 256, 256, 0, stream>>>(topk_idx, counts);
    k_route_scan   <<<1, 32, 0, stream>>>(counts, offsets, cursor);
    k_route_scatter<<<NASSIGN / 256, 256, 0, stream>>>(topk_idx, cursor, tok_idx, pos_of);

    dim3 g1(INTER / BN, NASSIGN / BM, NUM_EXPERTS);
    k_gemm1<<<g1, NTHREADS, 0, stream>>>(tokens, w1, b1, offsets, tok_idx, h_hi, h_lo);

    dim3 g2(HIDDEN / BN, NASSIGN / BM, NUM_EXPERTS);
    k_gemm2<<<g2, NTHREADS, 0, stream>>>(h_hi, h_lo, w2, b2, offsets, out_pairs);

    k_combine<<<(BATCH * HIDDEN) / 256, 256, 0, stream>>>(out_pairs, pos_of, topk_w, out);
}
